// ContextQueryAttention_10239202033849
// MI455X (gfx1250) — matrix operand order repaired and checked by reading, not compiled, not run on hardware
//
#include <hip/hip_runtime.h>
#include <hip/hip_bf16.h>

typedef unsigned short us;
typedef unsigned int u32;
typedef __attribute__((ext_vector_type(4)))  unsigned int u32x4;
typedef __attribute__((__vector_size__(16))) int          i32x4;
typedef __attribute__((ext_vector_type(16))) __bf16      v16bf;
typedef __attribute__((ext_vector_type(8)))  float       v8f;

#define BB 64
#define CC 1024
#define QQ 128
#define HH 512
#define LDSPITCH 40   // 32 bf16 + 8 pad, keeps 16B alignment (80B rows)

union FragBF { v16bf v; u32x4 q[2]; };
union Half8  { u32x4 q; us s[8]; };

// ---- gfx1250 async global->LDS path (guarded; falls back to sync copy) ----
#if defined(__has_builtin)
#  if __has_builtin(__builtin_amdgcn_global_load_async_to_lds_b128) && \
      __has_builtin(__builtin_amdgcn_s_wait_asynccnt)
#    define USE_ASYNC 1
#  endif
#endif
#ifndef USE_ASYNC
#  define USE_ASYNC 0
#endif

__device__ __forceinline__ void cp16(const us* g, us* l) {
#if USE_ASYNC
    __builtin_amdgcn_global_load_async_to_lds_b128(
        (__attribute__((address_space(1))) i32x4*)g,
        (__attribute__((address_space(3))) i32x4*)l, 0, 0);
#else
    *(u32x4*)l = *(const u32x4*)g;
#endif
}
#if USE_ASYNC
#  define ASYNC_FENCE(n) __builtin_amdgcn_s_wait_asynccnt(n)
#else
#  define ASYNC_FENCE(n) ((void)0)
#endif

__device__ __forceinline__ us f2bf(float f) {
    u32 u = __float_as_uint(f);
    u32 r = u + 0x7FFFu + ((u >> 16) & 1u);
    return (us)(r >> 16);
}

__device__ __forceinline__ v8f vzero8() {
    v8f z;
    #pragma unroll
    for (int i = 0; i < 8; ++i) z[i] = 0.0f;
    return z;
}

// ---------------- prep kernels ----------------
__global__ void prep_query(const float* __restrict__ xq, const float* __restrict__ W0,
                           const float* __restrict__ W2, us* __restrict__ xq_bf,
                           us* __restrict__ xqw_bf, float* __restrict__ sub0) {
    int row = blockIdx.x;                      // b*QQ + q
    const float* x = xq + (size_t)row * HH;
    int tid = threadIdx.x;
    float acc = 0.0f;
    for (int h = tid; h < HH; h += 256) {
        float v = x[h];
        xq_bf [(size_t)row * HH + h] = f2bf(v);
        xqw_bf[(size_t)row * HH + h] = f2bf(v * W2[h]);
        acc += v * W0[h];
    }
    #pragma unroll
    for (int off = 16; off >= 1; off >>= 1) acc += __shfl_xor(acc, off, 32);
    __shared__ float red[8];
    int lane = tid & 31, wv = tid >> 5;
    if (lane == 0) red[wv] = acc;
    __syncthreads();
    if (tid == 0) {
        float s = 0.0f;
        #pragma unroll
        for (int j = 0; j < 8; ++j) s += red[j];
        sub0[row] = s;
    }
}

__global__ void prep_context(const float* __restrict__ xc, const float* __restrict__ W1,
                             us* __restrict__ xc_bf, float* __restrict__ sub1) {
    int row = blockIdx.x;                      // b*CC + c
    const float* x = xc + (size_t)row * HH;
    int tid = threadIdx.x;
    float acc = 0.0f;
    for (int h = tid; h < HH; h += 256) {
        float v = x[h];
        xc_bf[(size_t)row * HH + h] = f2bf(v);
        acc += v * W1[h];
    }
    #pragma unroll
    for (int off = 16; off >= 1; off >>= 1) acc += __shfl_xor(acc, off, 32);
    __shared__ float red[8];
    int lane = tid & 31, wv = tid >> 5;
    if (lane == 0) red[wv] = acc;
    __syncthreads();
    if (tid == 0) {
        float s = 0.0f;
        #pragma unroll
        for (int j = 0; j < 8; ++j) s += red[j];
        sub1[row] = s;
    }
}

// ---------------- fragment loaders ----------------
__device__ __forceinline__ void loadA(FragBF& f, const us* tile, int row0, int lane) {
    int m   = row0 + (lane & 15);
    int klo = (lane >> 4) * 8;
    f.q[0] = *(const __attribute__((address_space(3))) u32x4*)
             ((const __attribute__((address_space(3))) us*)tile + m * LDSPITCH + klo);
    f.q[1] = *(const __attribute__((address_space(3))) u32x4*)
             ((const __attribute__((address_space(3))) us*)tile + m * LDSPITCH + klo + 16);
}
__device__ __forceinline__ void loadB(FragBF& f, const us* tile, int row0, int lane) {
    int n    = row0 + (lane & 15);
    int koff = (lane >> 4) * 8;
    f.q[0] = *(const __attribute__((address_space(3))) u32x4*)
             ((const __attribute__((address_space(3))) us*)tile + n * LDSPITCH + koff);
    f.q[1] = *(const __attribute__((address_space(3))) u32x4*)
             ((const __attribute__((address_space(3))) us*)tile + n * LDSPITCH + koff + 16);
}

#define WMMA_BF16(ACC, A, B) \
    (ACC) = __builtin_amdgcn_wmma_f32_16x16x32_bf16(false, (A).v, false, (B).v, (short)0, (ACC), false, false)

// ---------------- S = Xc * (Xq*W2)^T + sub1 + sub0 + bias ----------------
// grid: (CC/128) * BB, block 256 (8 waves, 2x4 wave grid, each wave 64x32)
__global__ void gemm_S(const us* __restrict__ xc_bf, const us* __restrict__ xqw_bf,
                       const float* __restrict__ sub0, const float* __restrict__ sub1,
                       const float* __restrict__ bias, float* __restrict__ S) {
    int mblk = blockIdx.x & 7;
    int b    = blockIdx.x >> 3;
    int c0   = mblk * 128;
    __shared__ us At[2][128 * LDSPITCH];
    __shared__ us Bt[2][128 * LDSPITCH];
    int tid = threadIdx.x, lane = tid & 31, wv = tid >> 5;
    int wm = wv & 1, wn = wv >> 1;
    v8f acc[4][2];
    #pragma unroll
    for (int i = 0; i < 4; ++i)
        #pragma unroll
        for (int j = 0; j < 2; ++j) acc[i][j] = vzero8();

    const us* Ab = xc_bf + ((size_t)b * CC + c0) * HH;
    const us* Bb = xqw_bf + (size_t)b * QQ * HH;

    auto stage = [&](int k0, int buf) {
        #pragma unroll
        for (int t = 0; t < 2; ++t) {
            int i = tid + t * 256;
            int r = i >> 2, q4 = i & 3;
            cp16(&Ab[(size_t)r * HH + k0 + q4 * 8], &At[buf][r * LDSPITCH + q4 * 8]);
            cp16(&Bb[(size_t)r * HH + k0 + q4 * 8], &Bt[buf][r * LDSPITCH + q4 * 8]);
        }
    };
    stage(0, 0);
    for (int k0 = 0; k0 < HH; k0 += 32) {
        int cur = (k0 >> 5) & 1;
        if (k0 + 32 < HH) { stage(k0 + 32, cur ^ 1); ASYNC_FENCE(4); }
        else              { ASYNC_FENCE(0); }
        __syncthreads();                       // tile `cur` visible to all waves
        FragBF a[4], bf[2];
        #pragma unroll
        for (int mi = 0; mi < 4; ++mi) loadA(a[mi], At[cur], wm * 64 + mi * 16, lane);
        #pragma unroll
        for (int ni = 0; ni < 2; ++ni) loadB(bf[ni], Bt[cur], wn * 32 + ni * 16, lane);
        #pragma unroll
        for (int mi = 0; mi < 4; ++mi)
            #pragma unroll
            for (int ni = 0; ni < 2; ++ni) WMMA_BF16(acc[mi][ni], a[mi], bf[ni]);
        __syncthreads();                       // all waves done reading `cur`
    }
    float bv = bias[0];
    int half = lane >> 4, nn = lane & 15;
    #pragma unroll
    for (int mi = 0; mi < 4; ++mi)
        #pragma unroll
        for (int ni = 0; ni < 2; ++ni) {
            int gq = wn * 32 + ni * 16 + nn;
            float s0 = sub0[b * QQ + gq];
            #pragma unroll
            for (int r = 0; r < 8; ++r) {
                int gc = c0 + wm * 64 + mi * 16 + half * 8 + r;
                S[((size_t)b * CC + gc) * QQ + gq] =
                    acc[mi][ni][r] + sub1[b * CC + gc] + s0 + bv;
            }
        }
}

// ---------------- softmax over q (axis -1), bf16 out ----------------
__global__ void softmax_q(const float* __restrict__ S, const float* __restrict__ qmask,
                          us* __restrict__ Ssm) {
    int wv = threadIdx.x >> 5, lane = threadIdx.x & 31;
    size_t row = (size_t)blockIdx.x * 8 + wv;        // b*CC + c
    int b = (int)(row / CC);
    const float* p = S + row * QQ;
    float v[4], mx = -3.0e38f;
    #pragma unroll
    for (int i = 0; i < 4; ++i) {
        int q = lane + i * 32;
        v[i] = p[q] + (1.0f - qmask[b * QQ + q]) * -1.0e30f;
        mx = fmaxf(mx, v[i]);
    }
    #pragma unroll
    for (int off = 16; off >= 1; off >>= 1) mx = fmaxf(mx, __shfl_xor(mx, off, 32));
    float e[4], sum = 0.0f;
    #pragma unroll
    for (int i = 0; i < 4; ++i) { e[i] = __expf(v[i] - mx); sum += e[i]; }
    #pragma unroll
    for (int off = 16; off >= 1; off >>= 1) sum += __shfl_xor(sum, off, 32);
    float inv = 1.0f / sum;
    #pragma unroll
    for (int i = 0; i < 4; ++i) Ssm[row * QQ + lane + i * 32] = f2bf(e[i] * inv);
}

// ---------------- softmax over c (axis 1), write transposed bf16 ScT[b][q][c] ----------------
__global__ void softmax_c(const float* __restrict__ S, const float* __restrict__ cmask,
                          us* __restrict__ ScT) {
    int tx = threadIdx.x & 31, ty = threadIdx.x >> 5;
    int qg = blockIdx.x & 3, b = blockIdx.x >> 2;
    int q = qg * 32 + tx;
    const float* Sb = S + (size_t)b * CC * QQ;
    const float* cm = cmask + (size_t)b * CC;
    __shared__ float red[8][33];
    float mx = -3.0e38f;
    for (int c = ty; c < CC; c += 8)
        mx = fmaxf(mx, Sb[(size_t)c * QQ + q] + (1.0f - cm[c]) * -1.0e30f);
    red[ty][tx] = mx;
    __syncthreads();
    if (ty == 0) {
        float m = red[0][tx];
        #pragma unroll
        for (int j = 1; j < 8; ++j) m = fmaxf(m, red[j][tx]);
        red[0][tx] = m;
    }
    __syncthreads();
    mx = red[0][tx];
    __syncthreads();
    float sum = 0.0f;
    for (int c = ty; c < CC; c += 8)
        sum += __expf(Sb[(size_t)c * QQ + q] + (1.0f - cm[c]) * -1.0e30f - mx);
    red[ty][tx] = sum;
    __syncthreads();
    if (ty == 0) {
        float s = 0.0f;
        #pragma unroll
        for (int j = 0; j < 8; ++j) s += red[j][tx];
        red[0][tx] = s;
    }
    __syncthreads();
    float inv = 1.0f / red[0][tx];
    us* dst = ScT + ((size_t)b * QQ + q) * CC;
    for (int c = ty; c < CC; c += 8)
        dst[c] = f2bf(__expf(Sb[(size_t)c * QQ + q] + (1.0f - cm[c]) * -1.0e30f - mx) * inv);
}

// ---------------- tmp[b,q,h] = sum_c ScT[b,q,c] * Xc[b,c,h]  (bf16 out) ----------------
// grid: (HH/128) * BB, block 256
__global__ void gemm_tmp(const us* __restrict__ ScT, const us* __restrict__ xc_bf,
                         us* __restrict__ tmp_bf) {
    int nblk = blockIdx.x & 3;
    int b    = blockIdx.x >> 2;
    int n0   = nblk * 128;
    __shared__ us At[2][128 * LDSPITCH];
    __shared__ us Bt[2][128 * LDSPITCH];   // transposed: Bt[n][k] = Xc[k0+k][n0+n]
    int tid = threadIdx.x, lane = tid & 31, wv = tid >> 5;
    int wm = wv & 1, wn = wv >> 1;
    v8f acc[4][2];
    #pragma unroll
    for (int i = 0; i < 4; ++i)
        #pragma unroll
        for (int j = 0; j < 2; ++j) acc[i][j] = vzero8();

    const us* Ab = ScT + (size_t)b * QQ * CC;
    const us* Bs = xc_bf + (size_t)b * CC * HH;

    auto stageA = [&](int k0, int buf) {
        #pragma unroll
        for (int t = 0; t < 2; ++t) {
            int i = tid + t * 256;
            int r = i >> 2, q4 = i & 3;
            cp16(&Ab[(size_t)r * CC + k0 + q4 * 8], &At[buf][r * LDSPITCH + q4 * 8]);
        }
    };
    auto stageB = [&](int k0, int buf) {
        int k = tid & 31, ng = tid >> 5;
        Half8 d0, d1;
        d0.q = *(const u32x4*)&Bs[(size_t)(k0 + k) * HH + n0 + ng * 16];
        d1.q = *(const u32x4*)&Bs[(size_t)(k0 + k) * HH + n0 + ng * 16 + 8];
        #pragma unroll
        for (int j = 0; j < 8; ++j) Bt[buf][(ng * 16 + j) * LDSPITCH + k]     = d0.s[j];
        #pragma unroll
        for (int j = 0; j < 8; ++j) Bt[buf][(ng * 16 + 8 + j) * LDSPITCH + k] = d1.s[j];
    };
    stageA(0, 0); stageB(0, 0);
    for (int k0 = 0; k0 < CC; k0 += 32) {
        int cur = (k0 >> 5) & 1;
        if (k0 + 32 < CC) { stageA(k0 + 32, cur ^ 1); stageB(k0 + 32, cur ^ 1); ASYNC_FENCE(2); }
        else              { ASYNC_FENCE(0); }
        __syncthreads();
        FragBF a[4], bf[2];
        #pragma unroll
        for (int mi = 0; mi < 4; ++mi) loadA(a[mi], At[cur], wm * 64 + mi * 16, lane);
        #pragma unroll
        for (int ni = 0; ni < 2; ++ni) loadB(bf[ni], Bt[cur], wn * 32 + ni * 16, lane);
        #pragma unroll
        for (int mi = 0; mi < 4; ++mi)
            #pragma unroll
            for (int ni = 0; ni < 2; ++ni) WMMA_BF16(acc[mi][ni], a[mi], bf[ni]);
        __syncthreads();
    }
    int half = lane >> 4, nn = lane & 15;
    #pragma unroll
    for (int mi = 0; mi < 4; ++mi)
        #pragma unroll
        for (int ni = 0; ni < 2; ++ni) {
            int gh = n0 + wn * 32 + ni * 16 + nn;
            #pragma unroll
            for (int r = 0; r < 8; ++r) {
                int gq = wm * 64 + mi * 16 + half * 8 + r;
                tmp_bf[((size_t)b * QQ + gq) * HH + gh] = f2bf(acc[mi][ni][r]);
            }
        }
}

// ---------------- fused: c2q = S_ @ Xq ; q2c = S_ @ tmp ----------------
// grid: (CC/128)*(HH/128)*BB, block 256
__global__ void gemm_out(const us* __restrict__ Ssm, const us* __restrict__ xq_bf,
                         const us* __restrict__ tmp_bf, float* __restrict__ c2q,
                         float* __restrict__ q2c) {
    int blk  = blockIdx.x;
    int mblk = blk & 7;  blk >>= 3;
    int nblk = blk & 3;  blk >>= 2;
    int b    = blk;
    int c0 = mblk * 128, n0 = nblk * 128;
    __shared__ us At[2][128 * LDSPITCH];
    __shared__ us B1[2][128 * LDSPITCH];
    __shared__ us B2[2][128 * LDSPITCH];
    int tid = threadIdx.x, lane = tid & 31, wv = tid >> 5;
    int wm = wv & 1, wn = wv >> 1;
    v8f acc1[4][2], acc2[4][2];
    #pragma unroll
    for (int i = 0; i < 4; ++i)
        #pragma unroll
        for (int j = 0; j < 2; ++j) { acc1[i][j] = vzero8(); acc2[i][j] = vzero8(); }

    const us* Ab  = Ssm + ((size_t)b * CC + c0) * QQ;
    const us* B1s = xq_bf + (size_t)b * QQ * HH;
    const us* B2s = tmp_bf + (size_t)b * QQ * HH;

    auto stageA = [&](int k0, int buf) {
        #pragma unroll
        for (int t = 0; t < 2; ++t) {
            int i = tid + t * 256;
            int r = i >> 2, q4 = i & 3;
            cp16(&Ab[(size_t)r * QQ + k0 + q4 * 8], &At[buf][r * LDSPITCH + q4 * 8]);
        }
    };
    auto stageB = [&](int k0, int buf) {
        int k = tid & 31, ng = tid >> 5;
        Half8 d0, d1, e0, e1;
        d0.q = *(const u32x4*)&B1s[(size_t)(k0 + k) * HH + n0 + ng * 16];
        d1.q = *(const u32x4*)&B1s[(size_t)(k0 + k) * HH + n0 + ng * 16 + 8];
        e0.q = *(const u32x4*)&B2s[(size_t)(k0 + k) * HH + n0 + ng * 16];
        e1.q = *(const u32x4*)&B2s[(size_t)(k0 + k) * HH + n0 + ng * 16 + 8];
        #pragma unroll
        for (int j = 0; j < 8; ++j) {
            B1[buf][(ng * 16 + j) * LDSPITCH + k]     = d0.s[j];
            B1[buf][(ng * 16 + 8 + j) * LDSPITCH + k] = d1.s[j];
            B2[buf][(ng * 16 + j) * LDSPITCH + k]     = e0.s[j];
            B2[buf][(ng * 16 + 8 + j) * LDSPITCH + k] = e1.s[j];
        }
    };
    stageA(0, 0); stageB(0, 0);
    for (int k0 = 0; k0 < QQ; k0 += 32) {
        int cur = (k0 >> 5) & 1;
        if (k0 + 32 < QQ) { stageA(k0 + 32, cur ^ 1); stageB(k0 + 32, cur ^ 1); ASYNC_FENCE(2); }
        else              { ASYNC_FENCE(0); }
        __syncthreads();
        FragBF a[4], b1[2], b2[2];
        #pragma unroll
        for (int mi = 0; mi < 4; ++mi) loadA(a[mi], At[cur], wm * 64 + mi * 16, lane);
        #pragma unroll
        for (int ni = 0; ni < 2; ++ni) { loadB(b1[ni], B1[cur], wn * 32 + ni * 16, lane);
                                         loadB(b2[ni], B2[cur], wn * 32 + ni * 16, lane); }
        #pragma unroll
        for (int mi = 0; mi < 4; ++mi)
            #pragma unroll
            for (int ni = 0; ni < 2; ++ni) {
                WMMA_BF16(acc1[mi][ni], a[mi], b1[ni]);
                WMMA_BF16(acc2[mi][ni], a[mi], b2[ni]);
            }
        __syncthreads();
    }
    int half = lane >> 4, nn = lane & 15;
    #pragma unroll
    for (int mi = 0; mi < 4; ++mi)
        #pragma unroll
        for (int ni = 0; ni < 2; ++ni) {
            int gh = n0 + wn * 32 + ni * 16 + nn;
            #pragma unroll
            for (int r = 0; r < 8; ++r) {
                int gc = c0 + wm * 64 + mi * 16 + half * 8 + r;
                size_t idx = ((size_t)b * CC + gc) * HH + gh;
                __builtin_nontemporal_store(acc1[mi][ni][r], &c2q[idx]);
                __builtin_nontemporal_store(acc2[mi][ni][r], &q2c[idx]);
            }
        }
}

// ---------------- host launcher ----------------
extern "C" void kernel_launch(void* const* d_in, const int* in_sizes, int n_in,
                              void* d_out, int out_size, void* d_ws, size_t ws_size,
                              hipStream_t stream) {
    const float* x_context = (const float*)d_in[0];
    const float* x_query   = (const float*)d_in[1];
    const float* cmask     = (const float*)d_in[2];
    const float* qmask     = (const float*)d_in[3];
    const float* W0        = (const float*)d_in[4];
    const float* W1        = (const float*)d_in[5];
    const float* W2        = (const float*)d_in[6];
    const float* bias      = (const float*)d_in[7];

    char* ws = (char*)d_ws;
    size_t off = 0;
    us* xc_bf  = (us*)(ws + off); off += (size_t)BB * CC * HH * 2;   // 64MB
    us* xq_bf  = (us*)(ws + off); off += (size_t)BB * QQ * HH * 2;   // 8MB
    us* xqw_bf = (us*)(ws + off); off += (size_t)BB * QQ * HH * 2;   // 8MB
    us* tmp_bf = (us*)(ws + off); off += (size_t)BB * QQ * HH * 2;   // 8MB
    us* Ssm    = (us*)(ws + off); off += (size_t)BB * CC * QQ * 2;   // 16MB
    us* ScT    = (us*)(ws + off); off += (size_t)BB * QQ * CC * 2;   // 16MB
    float* S   = (float*)(ws + off); off += (size_t)BB * CC * QQ * 4; // 32MB
    float* sub0 = (float*)(ws + off); off += (size_t)BB * QQ * 4;
    float* sub1 = (float*)(ws + off); off += (size_t)BB * CC * 4;

    float* c2q = (float*)d_out;
    float* q2c = c2q + (size_t)BB * CC * HH;

    prep_query  <<<BB * QQ, 256, 0, stream>>>(x_query, W0, W2, xq_bf, xqw_bf, sub0);
    prep_context<<<BB * CC, 256, 0, stream>>>(x_context, W1, xc_bf, sub1);
    gemm_S      <<<BB * (CC / 128), 256, 0, stream>>>(xc_bf, xqw_bf, sub0, sub1, bias, S);
    softmax_q   <<<BB * CC / 8, 256, 0, stream>>>(S, qmask, Ssm);
    softmax_c   <<<BB * (QQ / 32), 256, 0, stream>>>(S, cmask, ScT);
    gemm_tmp    <<<BB * (HH / 128), 256, 0, stream>>>(ScT, xc_bf, tmp_bf);
    gemm_out    <<<BB * (CC / 128) * (HH / 128), 256, 0, stream>>>(Ssm, xq_bf, tmp_bf, c2q, q2c);
}
